// ProcessorNetwork_16441134809381
// MI455X (gfx1250) — matrix operand order repaired and checked by reading, not compiled, not run on hardware
//
#include <hip/hip_runtime.h>

#define BB 8
#define NN 2048
#define DD 128
#define FF 256        // 2*D
#define KSTEP 32
#define JW 64         // output rows (j) per workgroup
#define LROW 40       // slab row pitch in halves: 80 B = 5*16 -> every row 16B-aligned
#define CROW 264      // lcat row pitch in halves: 528 B = 33*16 -> 16B-aligned rows

typedef __attribute__((ext_vector_type(16))) _Float16 v16h;
typedef __attribute__((ext_vector_type(8)))  _Float16 h8;
typedef __attribute__((ext_vector_type(4)))  _Float16 h4;
typedef __attribute__((ext_vector_type(8)))  float    v8f;

// ---------------------------------------------------------------------------
// Kernel A: inv_deg[b*N + i] = 1 / count_nonzero(cost[b, i, :])
// ---------------------------------------------------------------------------
__global__ void degree_kernel(const float* __restrict__ C,
                              float* __restrict__ inv_deg) {
  int row = blockIdx.x * blockDim.x + threadIdx.x;   // 0 .. B*N-1
  const float4* p = (const float4*)(C + (size_t)row * NN);
  float cnt = 0.0f;
  for (int k = 0; k < NN / 4; ++k) {
    float4 v = p[k];
    cnt += (float)((v.x != 0.0f) + (v.y != 0.0f) + (v.z != 0.0f) + (v.w != 0.0f));
  }
  inv_deg[row] = 1.0f / cnt;
}

// ---------------------------------------------------------------------------
// Fused kernel: S = (C^T E) * inv_deg ; out = relu([E | S] @ W^T + bias)
// Grid: B * (N/64) workgroups of 128 threads (4 waves, wave32).
// Wave w owns output rows j0 + 16w .. j0 + 16w + 15, all 128 output cols.
// Double-buffered LDS slabs; fragments load as ds_load_b128 pairs.
// ---------------------------------------------------------------------------
__global__ __launch_bounds__(128)
void fused_kernel(const float* __restrict__ E,       // [B,N,D]
                  const float* __restrict__ C,       // [B,N,N]
                  const float* __restrict__ W,       // [D, 2D] row-major
                  const float* __restrict__ bias,    // [D]
                  const float* __restrict__ inv_deg, // [B,N]
                  float* __restrict__ out)           // [B,N,D]
{
  __shared__ __align__(16) _Float16 lA[2][JW][LROW];   // C^T slab: [j_local][k_local]
  __shared__ __align__(16) _Float16 lK[2][DD][LROW];   // E^T slab (GEMM1) / W slab (GEMM2)
  __shared__ __align__(16) _Float16 lcat[JW][CROW];    // concat rows [j_local][f]

  const int tid  = threadIdx.x;
  const int lane = tid & 31;
  const int wid  = tid >> 5;
  const int bwg  = NN / JW;                 // 32 workgroups per batch
  const int b    = blockIdx.x / bwg;
  const int j0   = (blockIdx.x % bwg) * JW;

  const int m   = lane & 15;                // tile column / A-row
  const int hi  = lane >> 4;                // lane half
  const int hi8 = hi * 8;                   // A-frag K offset & C/D M offset
  const int kb  = hi * 16;                  // unused; B-frag K offset is hi8 too
  const int jl  = wid * 16 + m;             // this lane's A-fragment local row

  const float* Cb = C + (size_t)b * NN * NN;
  const float* Eb = E + (size_t)b * NN * DD;

  // Stage C[b, k0+i, j0+j] -> lA[s][j][i] and E[b, k0+i, d] -> lK[s][d][i]
  auto stageCE = [&](int k0, int s) {
#pragma unroll
    for (int r = 0; r < 4; ++r) {           // 32x64 floats = 512 float4
      int e  = tid + 128 * r;
      int i  = e >> 4;
      int j4 = (e & 15) << 2;
      float4 v = *(const float4*)&Cb[(size_t)(k0 + i) * NN + (j0 + j4)];
      lA[s][j4 + 0][i] = (_Float16)v.x;
      lA[s][j4 + 1][i] = (_Float16)v.y;
      lA[s][j4 + 2][i] = (_Float16)v.z;
      lA[s][j4 + 3][i] = (_Float16)v.w;
    }
#pragma unroll
    for (int r = 0; r < 8; ++r) {           // 32x128 floats = 1024 float4
      int e  = tid + 128 * r;
      int i  = e >> 5;
      int d4 = (e & 31) << 2;
      float4 v = *(const float4*)&Eb[(size_t)(k0 + i) * DD + d4];
      lK[s][d4 + 0][i] = (_Float16)v.x;
      lK[s][d4 + 1][i] = (_Float16)v.y;
      lK[s][d4 + 2][i] = (_Float16)v.z;
      lK[s][d4 + 3][i] = (_Float16)v.w;
    }
  };

  v8f acc[8] = {};

  // ---------------- GEMM1: S_unnorm[j, d] = sum_i C[i, j] * E[i, d] ---------
  stageCE(0, 0);
  __syncthreads();
  const int NIT = NN / KSTEP;               // 64
  for (int it = 0; it < NIT; ++it) {
    const int cur = it & 1;
    if (it + 1 < NIT) stageCE((it + 1) * KSTEP, cur ^ 1);

    // A fragment: two 16B-aligned runs of 8 halves -> 2x ds_load_b128
    h8 a0 = *(const h8*)&lA[cur][jl][hi8];
    h8 a1 = *(const h8*)&lA[cur][jl][16 + hi8];
    v16h a = __builtin_shufflevector(a0, a1, 0, 1, 2, 3, 4, 5, 6, 7,
                                     8, 9, 10, 11, 12, 13, 14, 15);
#pragma unroll
    for (int t = 0; t < 8; ++t) {
      int n = t * 16 + m;
      // B fragment: two 16B-aligned runs of 8 halves -> 2x ds_load_b128
      h8 b0 = *(const h8*)&lK[cur][n][hi8];
      h8 b1 = *(const h8*)&lK[cur][n][16 + hi8];
      v16h bf = __builtin_shufflevector(b0, b1, 0, 1, 2, 3, 4, 5, 6, 7,
                                        8, 9, 10, 11, 12, 13, 14, 15);
      acc[t] = __builtin_amdgcn_wmma_f32_16x16x32_f16(
          false, a, false, bf, (short)0, acc[t], false, false);
    }
    __syncthreads();
  }

  // -------- normalize, build concat = [E | S] rows in LDS (f16) -------------
#pragma unroll
  for (int r = 0; r < 8; ++r) {
    int jrow = j0 + wid * 16 + hi8 + r;      // C/D layout: VGPR r <-> M = hi*8 + r
    float idg = inv_deg[b * NN + jrow];
#pragma unroll
    for (int t = 0; t < 8; ++t) {
      float s = acc[t][r] * idg;
      lcat[wid * 16 + hi8 + r][DD + t * 16 + m] = (_Float16)s;
    }
  }
#pragma unroll
  for (int r = 0; r < 16; ++r) {             // 64x128 floats = 2048 float4
    int e  = tid + 128 * r;
    int j  = e >> 5;
    int d4 = (e & 31) << 2;
    float4 v = *(const float4*)&Eb[(size_t)(j0 + j) * DD + d4];
    h4 hv;
    hv.x = (_Float16)v.x; hv.y = (_Float16)v.y;
    hv.z = (_Float16)v.z; hv.w = (_Float16)v.w;
    *(h4*)&lcat[j][d4] = hv;                 // 8B-aligned ds_store_b64
  }

  // Stage W[n, k0+k] -> lK[s][n][k] (128x32 floats = 1024 float4)
  auto stageW = [&](int k0, int s) {
#pragma unroll
    for (int r = 0; r < 8; ++r) {
      int e  = tid + 128 * r;
      int n  = e >> 3;
      int k4 = (e & 7) << 2;
      float4 v = *(const float4*)&W[(size_t)n * FF + (k0 + k4)];
      lK[s][n][k4 + 0] = (_Float16)v.x;
      lK[s][n][k4 + 1] = (_Float16)v.y;
      lK[s][n][k4 + 2] = (_Float16)v.z;
      lK[s][n][k4 + 3] = (_Float16)v.w;
    }
  };

  stageW(0, 0);
  __syncthreads();

  // ---------------- GEMM2: out[j, n] = relu(cat[j, :] . W[n, :] + bias) -----
  v8f acc2[8] = {};
  const int NIT2 = FF / KSTEP;               // 8
  for (int it = 0; it < NIT2; ++it) {
    const int cur = it & 1;
    const int k0  = it * KSTEP;
    if (it + 1 < NIT2) stageW((it + 1) * KSTEP, cur ^ 1);

    h8 a0 = *(const h8*)&lcat[jl][k0 + hi8];
    h8 a1 = *(const h8*)&lcat[jl][k0 + 16 + hi8];
    v16h a = __builtin_shufflevector(a0, a1, 0, 1, 2, 3, 4, 5, 6, 7,
                                     8, 9, 10, 11, 12, 13, 14, 15);
#pragma unroll
    for (int t = 0; t < 8; ++t) {
      int n = t * 16 + m;
      h8 b0 = *(const h8*)&lK[cur][n][hi8];
      h8 b1 = *(const h8*)&lK[cur][n][16 + hi8];
      v16h bf = __builtin_shufflevector(b0, b1, 0, 1, 2, 3, 4, 5, 6, 7,
                                        8, 9, 10, 11, 12, 13, 14, 15);
      acc2[t] = __builtin_amdgcn_wmma_f32_16x16x32_f16(
          false, a, false, bf, (short)0, acc2[t], false, false);
    }
    __syncthreads();
  }

  // ---------------- epilogue: bias + relu + store f32 -----------------------
  float bcol[8];
#pragma unroll
  for (int t = 0; t < 8; ++t) bcol[t] = bias[t * 16 + m];

  float* Ob = out + (size_t)b * NN * DD;
#pragma unroll
  for (int r = 0; r < 8; ++r) {
    int jrow = j0 + wid * 16 + hi8 + r;
#pragma unroll
    for (int t = 0; t < 8; ++t) {
      float v = acc2[t][r] + bcol[t];
      Ob[(size_t)jrow * DD + t * 16 + m] = v > 0.0f ? v : 0.0f;
    }
  }
}

// ---------------------------------------------------------------------------
extern "C" void kernel_launch(void* const* d_in, const int* in_sizes, int n_in,
                              void* d_out, int out_size, void* d_ws, size_t ws_size,
                              hipStream_t stream) {
  (void)in_sizes; (void)n_in; (void)out_size; (void)ws_size;
  const float* E    = (const float*)d_in[0];   // embeddings [B,N,D]
  const float* C    = (const float*)d_in[1];   // cost_matrix [B,N,N]
  const float* W    = (const float*)d_in[2];   // W [D, 2D]
  const float* bias = (const float*)d_in[3];   // b [D]
  float* inv_deg    = (float*)d_ws;            // [B*N] floats (64 KB)

  degree_kernel<<<(BB * NN) / 256, 256, 0, stream>>>(C, inv_deg);
  fused_kernel<<<BB * (NN / JW), 128, 0, stream>>>(E, C, W, bias, inv_deg,
                                                   (float*)d_out);
}
